// BinaryModel_60000693125684
// MI455X (gfx1250) — 16-bit and 8-bit matrix operand order repaired and checked by reading, compiled, not run on hardware
//
#include <hip/hip_runtime.h>
#include <hip/hip_bf16.h>


typedef __attribute__((ext_vector_type(16))) _Float16 v16h;
typedef __attribute__((ext_vector_type(8)))  float    v8f;
typedef __attribute__((ext_vector_type(8)))  int      v8i;
typedef __attribute__((ext_vector_type(4)))  int      v4i;
typedef __attribute__((ext_vector_type(2)))  int      v2i;
typedef __attribute__((ext_vector_type(4)))  float    v4f;

#define IMG_H 64
#define IMG_W 64
#define CIN1  32
#define COUT  64

__device__ __forceinline__ void async_load_b128(unsigned lds_off, const void* gaddr) {
    asm volatile("global_load_async_to_lds_b128 %0, %1, off"
                 :: "v"(lds_off), "v"(gaddr) : "memory");
}
__device__ __forceinline__ void wait_asynccnt0() {
    asm volatile("s_wait_asynccnt 0x0" ::: "memory");
}

__global__ void prep_kernel(const float* __restrict__ w1, const float* __restrict__ b1,
                            const float* __restrict__ w2, const float* __restrict__ b2,
                            unsigned* __restrict__ w1p, float* __restrict__ b1s,
                            unsigned* __restrict__ w2p, int* __restrict__ b2s)
{
    int tid  = blockIdx.x * blockDim.x + threadIdx.x;
    int nthr = gridDim.x * blockDim.x;

    for (int idx = tid; idx < 9216; idx += nthr) {
        int v    = idx & 7;
        int lane = (idx >> 3) & 31;
        int ot   = (idx >> 8) & 3;
        int t    = idx >> 10;
        int nn   = lane & 15;
        int g    = lane >> 4;
        int ky   = t / 3, kx = t % 3;
        int o    = ot * 16 + nn;
        int k0   = (v < 4) ? (g * 8 + 2 * v) : (16 + g * 8 + 2 * (v - 4));
        float a0 = w1[((o * CIN1 + k0    ) * 3 + ky) * 3 + kx];
        float a1 = w1[((o * CIN1 + k0 + 1) * 3 + ky) * 3 + kx];
        union { _Float16 h[2]; unsigned u; } p;
        p.h[0] = (a0 >= 0.f) ? (_Float16)1.f : (_Float16)-1.f;
        p.h[1] = (a1 >= 0.f) ? (_Float16)1.f : (_Float16)-1.f;
        w1p[idx] = p.u;
    }

    for (int idx = tid; idx < 9216; idx += nthr) {
        int v    = idx & 7;
        int lane = (idx >> 3) & 31;
        int ot   = (idx >> 8) & 3;
        int t    = idx >> 10;
        int nn   = lane & 15;
        int g    = lane >> 4;
        int ky   = t / 3, kx = t % 3;
        int o    = ot * 16 + nn;
        int kb   = (v < 4) ? (g * 16 + 4 * v) : (32 + g * 16 + 4 * (v - 4));
        union { signed char b[4]; unsigned u; } p;
#pragma unroll
        for (int bb = 0; bb < 4; ++bb) {
            float a = w2[((o * COUT + (kb + bb)) * 3 + ky) * 3 + kx];
            p.b[bb] = (a >= 0.f) ? (signed char)1 : (signed char)-1;
        }
        w2p[idx] = p.u;
    }

    for (int idx = tid; idx < 64; idx += nthr) {
        b1s[idx] = (b1[idx] >= 0.f) ? 1.f : -1.f;
        b2s[idx] = (b2[idx] >= 0.f) ? 1 : -1;
    }
}

__global__ __launch_bounds__(256) void conv1_kernel(const float* __restrict__ x,
                                                    const unsigned* __restrict__ w1p,
                                                    const float* __restrict__ b1s,
                                                    signed char* __restrict__ hbuf)
{
    __shared__ __attribute__((aligned(16))) _Float16    xin[6 * 66 * CIN1];
    __shared__ __attribute__((aligned(16))) unsigned    wlds[9216];
    __shared__ __attribute__((aligned(16))) signed char hstage[16 * 16 * 64];

    int tid = threadIdx.x;
    int bid = blockIdx.x;
    int n   = bid >> 4;
    int y0  = (bid & 15) << 2;
    const float* xn = x + (size_t)n * CIN1 * IMG_H * IMG_W;

    for (int i = tid; i < 2304; i += 256)
        async_load_b128((unsigned)(size_t)&wlds[i * 4], w1p + i * 4);

    for (int idx = tid; idx < 6 * CIN1 * 66; idx += 256) {
        int c  = idx % 66;
        int rc = idx / 66;
        int r  = rc % 6;
        int ch = rc / 6;
        int gy = y0 - 1 + r;
        int gx = c - 1;
        float v = 0.f;
        if (gy >= 0 && gy < IMG_H && gx >= 0 && gx < IMG_W)
            v = xn[ch * IMG_H * IMG_W + gy * IMG_W + gx];
        xin[(r * 66 + c) * CIN1 + ch] = (_Float16)v;
    }
    wait_asynccnt0();
    __syncthreads();

    int lane    = tid & 31;
    int wave    = tid >> 5;
    int r0      = (wave >> 2) << 1;
    int colbase = (wave & 3) << 4;
    int m       = lane & 15;
    int g       = lane >> 4;

    v8f acc[2][4];
#pragma unroll
    for (int ot = 0; ot < 4; ++ot) {
        float bv = b1s[ot * 16 + m];
#pragma unroll
        for (int e = 0; e < 8; ++e) { acc[0][ot][e] = bv; acc[1][ot][e] = bv; }
    }

#pragma unroll
    for (int t = 0; t < 9; ++t) {
        int ky = t / 3, kx = t % 3;
        int c  = colbase + m + kx;
        union { v16h h; v4i q[2]; } A0, A1;
        {
            const _Float16* ap = &xin[((r0 + 0 + ky) * 66 + c) * CIN1];
            A0.q[0] = *(const v4i*)(ap + g * 8);
            A0.q[1] = *(const v4i*)(ap + 16 + g * 8);
        }
        {
            const _Float16* ap = &xin[((r0 + 1 + ky) * 66 + c) * CIN1];
            A1.q[0] = *(const v4i*)(ap + g * 8);
            A1.q[1] = *(const v4i*)(ap + 16 + g * 8);
        }
#pragma unroll
        for (int ot = 0; ot < 4; ++ot) {
            const unsigned* bp = &wlds[((t * 4 + ot) * 32 + lane) * 8];
            union { v16h h; v4i q[2]; } B;
            B.q[0] = *(const v4i*)(bp);
            B.q[1] = *(const v4i*)(bp + 4);
            acc[0][ot] = __builtin_amdgcn_wmma_f32_16x16x32_f16(
                false, A0.h, false, B.h, (short)0, acc[0][ot], false, false);
            acc[1][ot] = __builtin_amdgcn_wmma_f32_16x16x32_f16(
                false, A1.h, false, B.h, (short)0, acc[1][ot], false, false);
        }
    }

#pragma unroll
    for (int j = 0; j < 2; ++j) {
        signed char* hs = &hstage[(wave * 2 + j) * 16 * 64];
#pragma unroll
        for (int ot = 0; ot < 4; ++ot) {
            int ch = ot * 16 + m;
#pragma unroll
            for (int r8 = 0; r8 < 8; ++r8) {
                int M = r8 + 8 * g;
                hs[M * 64 + ch] = (acc[j][ot][r8] >= 0.f) ? (signed char)1 : (signed char)-1;
            }
        }
        size_t gbase = (((size_t)n * IMG_H + (y0 + r0 + j)) * IMG_W + colbase) * 64;
        const v4i* src = (const v4i*)hs;
        v4i d0 = src[lane * 2 + 0];
        v4i d1 = src[lane * 2 + 1];
        *(v4i*)(hbuf + gbase + lane * 32 + 0)  = d0;
        *(v4i*)(hbuf + gbase + lane * 32 + 16) = d1;
    }
}

__global__ __launch_bounds__(256) void conv2_kernel(const signed char* __restrict__ hbuf,
                                                    const unsigned* __restrict__ w2p,
                                                    const int* __restrict__ b2s,
                                                    const float* __restrict__ scale,
                                                    float* __restrict__ out)
{
    __shared__ __attribute__((aligned(16))) signed char hin[6 * 66 * COUT];
    __shared__ __attribute__((aligned(16))) unsigned    wlds[9216];
    __shared__ __attribute__((aligned(16))) float       ostage[2 * 64 * 64];

    int tid = threadIdx.x;
    int bid = blockIdx.x;
    int n   = bid >> 4;
    int y0  = (bid & 15) << 2;

    for (int i = tid; i < 2304; i += 256)
        async_load_b128((unsigned)(size_t)&wlds[i * 4], w2p + i * 4);

    for (int idx = tid; idx < 6 * 66 * 4; idx += 256) {
        int q  = idx & 3;
        int rc = idx >> 2;
        int c  = rc % 66;
        int r  = rc / 66;
        int gy = y0 - 1 + r;
        int gx = c - 1;
        unsigned loff = (unsigned)(size_t)&hin[(r * 66 + c) * 64 + q * 16];
        if (gy >= 0 && gy < IMG_H && gx >= 0 && gx < IMG_W) {
            const signed char* gp =
                hbuf + (((size_t)n * IMG_H + gy) * IMG_W + gx) * 64 + q * 16;
            async_load_b128(loff, gp);
        } else {
            *(v4i*)(&hin[(r * 66 + c) * 64 + q * 16]) = (v4i){0, 0, 0, 0};
        }
    }
    wait_asynccnt0();
    __syncthreads();

    int lane    = tid & 31;
    int wave    = tid >> 5;
    int r0      = (wave >> 2) << 1;
    int colbase = (wave & 3) << 4;
    int m       = lane & 15;
    int g       = lane >> 4;

    v8i acc[2][4];
#pragma unroll
    for (int ot = 0; ot < 4; ++ot) {
        int bv = b2s[ot * 16 + m];
#pragma unroll
        for (int e = 0; e < 8; ++e) { acc[0][ot][e] = bv; acc[1][ot][e] = bv; }
    }

#pragma unroll
    for (int t = 0; t < 9; ++t) {
        int ky = t / 3, kx = t % 3;
        int c  = colbase + m + kx;
        union { v8i v; v2i d[4]; } A0, A1;
        {
            const signed char* ap = &hin[((r0 + 0 + ky) * 66 + c) * 64];
            A0.d[0] = *(const v2i*)(ap + 0  + g * 16);
            A0.d[1] = *(const v2i*)(ap + 8  + g * 16);
            A0.d[2] = *(const v2i*)(ap + 32 + g * 16);
            A0.d[3] = *(const v2i*)(ap + 40 + g * 16);
        }
        {
            const signed char* ap = &hin[((r0 + 1 + ky) * 66 + c) * 64];
            A1.d[0] = *(const v2i*)(ap + 0  + g * 16);
            A1.d[1] = *(const v2i*)(ap + 8  + g * 16);
            A1.d[2] = *(const v2i*)(ap + 32 + g * 16);
            A1.d[3] = *(const v2i*)(ap + 40 + g * 16);
        }
#pragma unroll
        for (int ot = 0; ot < 4; ++ot) {
            const unsigned* bp = &wlds[((t * 4 + ot) * 32 + lane) * 8];
            union { v8i v; v4i q[2]; } B;
            B.q[0] = *(const v4i*)(bp);
            B.q[1] = *(const v4i*)(bp + 4);
            acc[0][ot] = __builtin_amdgcn_wmma_i32_16x16x64_iu8(
                true, A0.v, true, B.v, acc[0][ot], false, false);
            acc[1][ot] = __builtin_amdgcn_wmma_i32_16x16x64_iu8(
                true, A1.v, true, B.v, acc[1][ot], false, false);
        }
    }

    float s = scale[0];
    int slot = wave >> 2;
#pragma unroll
    for (int p = 0; p < 2; ++p) {
#pragma unroll
        for (int ot = 0; ot < 4; ++ot) {
            int ch = ot * 16 + m;
#pragma unroll
            for (int r8 = 0; r8 < 8; ++r8) {
                int M   = r8 + 8 * g;
                int col = colbase + M;
                ostage[(slot * 64 + ch) * 64 + col] = s * (float)acc[p][ot][r8];
            }
        }
        __syncthreads();
        float* outn = out + (size_t)n * COUT * IMG_H * IMG_W;
        for (int idx = tid; idx < 2048; idx += 256) {
            int cc     = idx & 15;
            int chslot = idx >> 4;
            int ch     = chslot & 63;
            int sl     = chslot >> 6;
            v4f v = *(const v4f*)(&ostage[chslot * 64 + cc * 4]);
            *(v4f*)(outn + (size_t)ch * IMG_H * IMG_W +
                    (size_t)(y0 + sl * 2 + p) * IMG_W + cc * 4) = v;
        }
        __syncthreads();
    }
}

extern "C" void kernel_launch(void* const* d_in, const int* in_sizes, int n_in,
                              void* d_out, int out_size, void* d_ws, size_t ws_size,
                              hipStream_t stream)
{
    (void)in_sizes; (void)n_in; (void)out_size; (void)ws_size;

    const float* x     = (const float*)d_in[0];
    const float* w1    = (const float*)d_in[1];
    const float* b1    = (const float*)d_in[2];
    const float* w2    = (const float*)d_in[3];
    const float* b2    = (const float*)d_in[4];
    const float* scale = (const float*)d_in[5];

    char*        ws   = (char*)d_ws;
    unsigned*    w1p  = (unsigned*)(ws);
    unsigned*    w2p  = (unsigned*)(ws + 36864);
    float*       b1s  = (float*)(ws + 73728);
    int*         b2s  = (int*)(ws + 73984);
    signed char* hbuf = (signed char*)(ws + 74240);

    prep_kernel<<<dim3(36), dim3(256), 0, stream>>>(w1, b1, w2, b2, w1p, b1s, w2p, b2s);
    conv1_kernel<<<dim3(1024), dim3(256), 0, stream>>>(x, w1p, b1s, hbuf);
    conv2_kernel<<<dim3(1024), dim3(256), 0, stream>>>(hbuf, w2p, b2s, scale, (float*)d_out);
}
